// AMGAN_68040871903544
// MI455X (gfx1250) — compile-verified, matrix operand order checked by reading, not run on hardware
//
// AMGAN attention pipeline for MI455X (gfx1250, wave32, WMMA).
// Compute-bound (~20 GFLOP vs ~3 MB traffic) -> everything matmul-shaped goes
// through v_wmma_f32_16x16x32_f16 (f16 in, f32 accumulate). Head dim 32 == WMMA K.
// This revision: vectorized b128 fragment loads, 16x64 GEMM wave tiles (4 WMMA
// per K-step sharing one A fragment), ds_load_b128 P-tile reads, prefetch.
#include <hip/hip_runtime.h>
#include <hip/hip_fp16.h>
#include <math.h>

typedef __attribute__((ext_vector_type(16))) _Float16 v16h;
typedef __attribute__((ext_vector_type(8)))  _Float16 v8h;
typedef __attribute__((ext_vector_type(8)))  float    v8f;
typedef __attribute__((ext_vector_type(4)))  float    v4f;

#define NG 4096
#define ND 2048
#define DMODEL 128
#define NH 4
#define HD 32
#define LCCS 0.05f   // 0.5 * LCC_SCALE

// ---------------------------------------------------------------- utilities
__device__ __forceinline__ v8f wmma16(v16h a, v16h b, v8f c) {
  return __builtin_amdgcn_wmma_f32_16x16x32_f16(false, a, false, b, (short)0, c, false, false);
}
__device__ __forceinline__ v16h make16(v8h lo, v8h hi) {
  return __builtin_shufflevector(lo, hi, 0, 1, 2, 3, 4, 5, 6, 7, 8, 9, 10, 11, 12, 13, 14, 15);
}
__device__ __forceinline__ v8h cvt8(v4f x, v4f y) {
  v8h r;
  r[0] = (_Float16)x[0]; r[1] = (_Float16)x[1]; r[2] = (_Float16)x[2]; r[3] = (_Float16)x[3];
  r[4] = (_Float16)y[0]; r[5] = (_Float16)y[1]; r[6] = (_Float16)y[2]; r[7] = (_Float16)y[3];
  return r;
}
__device__ __forceinline__ float red16_max(float v) {
  v = fmaxf(v, __shfl_xor(v, 1, 32));
  v = fmaxf(v, __shfl_xor(v, 2, 32));
  v = fmaxf(v, __shfl_xor(v, 4, 32));
  v = fmaxf(v, __shfl_xor(v, 8, 32));
  return v;
}
__device__ __forceinline__ float red16_sum(float v) {
  v += __shfl_xor(v, 1, 32);
  v += __shfl_xor(v, 2, 32);
  v += __shfl_xor(v, 4, 32);
  v += __shfl_xor(v, 8, 32);
  return v;
}

// ---------------------------------------------------------- small kernels
__global__ void k_pos_add(const float* __restrict__ E, float* __restrict__ x, int S) {
  int idx = blockIdx.x * blockDim.x + threadIdx.x;
  if (idx >= S * DMODEL) return;
  int s = idx >> 7, d = idx & 127;
  float f = __expf(-(float)(d & ~1) * (logf(10000.f) / (float)DMODEL));
  float ang = (float)s * f;
  float pe = (d & 1) ? __cosf(ang) : __sinf(ang);
  x[idx] = E[idx] + pe;
}

// WT[n][k] = (f16) W[k][n]
__global__ void k_w2h_t(const float* __restrict__ W, _Float16* __restrict__ WT, int K, int N) {
  int idx = blockIdx.x * blockDim.x + threadIdx.x;
  if (idx >= K * N) return;
  int n = idx / K, k = idx - n * K;
  WT[idx] = (_Float16)W[(size_t)k * N + n];
}

// LayerNorm over last dim (=128), optional sigmoid. grid = S, block = 128.
__global__ void k_layernorm(const float* __restrict__ x, const float* __restrict__ g,
                            const float* __restrict__ b, float* __restrict__ y, int do_sig) {
  __shared__ float sbuf[DMODEL];
  int s = blockIdx.x, d = threadIdx.x;
  float v = x[(size_t)s * DMODEL + d];
  sbuf[d] = v;
  __syncthreads();
  for (int off = 64; off > 0; off >>= 1) { if (d < off) sbuf[d] += sbuf[d + off]; __syncthreads(); }
  float mean = sbuf[0] * (1.f / DMODEL);
  __syncthreads();
  float c = v - mean;
  sbuf[d] = c * c;
  __syncthreads();
  for (int off = 64; off > 0; off >>= 1) { if (d < off) sbuf[d] += sbuf[d + off]; __syncthreads(); }
  float var = sbuf[0] * (1.f / DMODEL);
  float out = c * rsqrtf(var + 1e-5f) * g[d] + b[d];
  if (do_sig) out = 1.f / (1.f + __expf(-out));
  y[(size_t)s * DMODEL + d] = out;
}

// Per-head relayout to f16. grid = S, block = 128 (wave = head, lane = dim).
// mode 0: l2norm + hd^-0.5 (q) -> [h][s][hd]; 1: l2norm (k) -> [h][s][hd];
// mode 2: transpose (v) -> [h][hd][s]
__global__ void k_head_relayout(const float* __restrict__ in, int ld, _Float16* __restrict__ out,
                                int S, int mode) {
  int s = blockIdx.x;
  int head = threadIdx.x >> 5;
  int lane = threadIdx.x & 31;
  float v = in[(size_t)s * ld + head * HD + lane];
  if (mode != 2) {
    float ss = v * v;
    for (int off = 16; off > 0; off >>= 1) ss += __shfl_xor(ss, off, 32);
    v = v / fmaxf(sqrtf(ss), 1e-12f);
    if (mode == 0) v *= 0.17677669529663687f;  // 32^-0.5
    out[((size_t)head * S + s) * HD + lane] = (_Float16)v;
  } else {
    out[((size_t)head * HD + lane) * S + s] = (_Float16)v;
  }
}

__global__ void k_concat(const float* __restrict__ a, const float* __restrict__ b,
                         float* __restrict__ c, int S) {
  int idx = blockIdx.x * blockDim.x + threadIdx.x;
  if (idx >= S * 2 * DMODEL) return;
  int s = idx >> 8, d = idx & 255;
  c[idx] = (d < DMODEL) ? a[(size_t)s * DMODEL + d] : b[(size_t)s * DMODEL + d - DMODEL];
}

__global__ void k_gated_out(const float* __restrict__ x, const float* __restrict__ gate,
                            const float* __restrict__ proj, float* __restrict__ out, int n) {
  int idx = blockIdx.x * blockDim.x + threadIdx.x;
  if (idx >= n) return;
  out[idx] = x[idx] + gate[idx] * proj[idx];
}

// ------------------------------------------------------------- WMMA GEMM
// C[M,N] = act(A[M,K](f32) * WT[N,K](f16) + bias[N]) (+ residual)
// One wave per 16x64 output tile: one A fragment feeds 4 B fragments / 4
// accumulators per 32-wide K step (4 WMMA per iteration, 4x A reuse).
// grid = (M/16, N/64), block = 32. K, N multiples of 32 / 64.
__global__ void k_gemm_wmma(const float* __restrict__ A, const _Float16* __restrict__ WT,
                            const float* __restrict__ bias, const float* __restrict__ residual,
                            float* __restrict__ C, int M, int N, int K, int act) {
  const int lane = threadIdx.x;
  const int tm = blockIdx.x << 4;
  const int tn = blockIdx.y << 6;
  const int l15 = lane & 15;
  const int half = lane >> 4;
  const int khalf = half << 3;
  const float* arow = A + (size_t)(tm + l15) * K;
  const _Float16* brow0 = WT + (size_t)(tn + l15) * K;
  const _Float16* brow1 = brow0 + (size_t)16 * K;
  const _Float16* brow2 = brow0 + (size_t)32 * K;
  const _Float16* brow3 = brow0 + (size_t)48 * K;
  v8f acc0 = {}, acc1 = {}, acc2 = {}, acc3 = {};
  for (int k0 = 0; k0 < K; k0 += 32) {
    v4f a0 = *(const v4f*)(arow + k0 + khalf);
    v4f a1 = *(const v4f*)(arow + k0 + khalf + 4);
    v4f a2 = *(const v4f*)(arow + k0 + 16 + khalf);
    v4f a3 = *(const v4f*)(arow + k0 + 16 + khalf + 4);
    v16h a = make16(cvt8(a0, a1), cvt8(a2, a3));
    v16h b0 = make16(*(const v8h*)(brow0 + k0 + khalf), *(const v8h*)(brow0 + k0 + 16 + khalf));
    v16h b1 = make16(*(const v8h*)(brow1 + k0 + khalf), *(const v8h*)(brow1 + k0 + 16 + khalf));
    v16h b2 = make16(*(const v8h*)(brow2 + k0 + khalf), *(const v8h*)(brow2 + k0 + 16 + khalf));
    v16h b3 = make16(*(const v8h*)(brow3 + k0 + khalf), *(const v8h*)(brow3 + k0 + 16 + khalf));
    acc0 = wmma16(a, b0, acc0);
    acc1 = wmma16(a, b1, acc1);
    acc2 = wmma16(a, b2, acc2);
    acc3 = wmma16(a, b3, acc3);
  }
  v8f accs[4] = {acc0, acc1, acc2, acc3};
#pragma unroll
  for (int t = 0; t < 4; ++t) {
    const int col = tn + 16 * t + l15;
    const float bv = bias ? bias[col] : 0.f;
#pragma unroll
    for (int r = 0; r < 8; ++r) {
      int row = tm + r + (half << 3);
      float v = accs[t][r] + bv;
      if (act == 1) v = 0.5f * v * (1.f + erff(v * 0.70710678118654752f));  // exact GELU
      if (residual) v += residual[(size_t)row * N + col];
      C[(size_t)row * N + col] = v;
    }
  }
}

// -------------------------------------------------------- flash attention
// Cosine-normalized heads (q,k pre-normalized in relayout), additive LCC bias,
// clamp +-10, online softmax. One wave per (16-query tile, head).
// qh[h][Sq][32] f16, kh[h][Skv][32] f16, vT[h][32][Skv] f16, out[Sq][128] f32.
__global__ void k_flash_attn(const _Float16* __restrict__ qh, const _Float16* __restrict__ kh,
                             const _Float16* __restrict__ vT, const float* __restrict__ lccq,
                             const float* __restrict__ lcck, float* __restrict__ out,
                             int Sq, int Skv) {
  __shared__ _Float16 pbuf[16][40];  // 80B row stride: 16B-aligned b128 LDS reads
  const int lane = threadIdx.x;
  const int qbase = blockIdx.x << 4;
  const int h = blockIdx.y;
  const int l15 = lane & 15;
  const int half = lane >> 4;
  const int khalf = half << 3;

  // Q A-fragment, loaded once (K dim = hd = 32 exactly): two b128 loads
  const _Float16* qrow = qh + ((size_t)h * Sq + qbase + l15) * HD;
  v16h qa = make16(*(const v8h*)(qrow + khalf), *(const v8h*)(qrow + 16 + khalf));

  float lq[8];
#pragma unroll
  for (int r = 0; r < 8; ++r) lq[r] = lccq[qbase + r + (half << 3)] * LCCS;

  v8f acc0 = {}, acc1 = {};
  float mrow[8], lrow[8];
#pragma unroll
  for (int r = 0; r < 8; ++r) { mrow[r] = -1e30f; lrow[r] = 0.f; }

  const _Float16* krow = kh + ((size_t)h * Skv + l15) * HD;
  const _Float16* vrow0 = vT + ((size_t)h * HD + l15) * Skv;
  const _Float16* vrow1 = vT + ((size_t)h * HD + 16 + l15) * Skv;

  for (int kv0 = 0; kv0 < Skv; kv0 += 32) {
    // K^T B-fragments for kv columns [kv0, kv0+16) and [kv0+16, kv0+32)
    const _Float16* kr0 = krow + (size_t)kv0 * HD;
    const _Float16* kr1 = kr0 + 16 * HD;
    v16h kb0 = make16(*(const v8h*)(kr0 + khalf), *(const v8h*)(kr0 + 16 + khalf));
    v16h kb1 = make16(*(const v8h*)(kr1 + khalf), *(const v8h*)(kr1 + 16 + khalf));
    if (kv0 + 32 < Skv) {  // prefetch next K/V tiles behind the score WMMAs
      __builtin_prefetch(kr0 + 32 * HD, 0, 0);
      __builtin_prefetch(vrow0 + kv0 + 32, 0, 0);
      __builtin_prefetch(vrow1 + kv0 + 32, 0, 0);
    }
    v8f s0 = {}, s1 = {};
    s0 = wmma16(qa, kb0, s0);
    s1 = wmma16(qa, kb1, s1);

    float lk0 = lcck[kv0 + l15] * LCCS;
    float lk1 = lcck[kv0 + 16 + l15] * LCCS;
#pragma unroll
    for (int r = 0; r < 8; ++r) {
      float a0 = fminf(fmaxf(s0[r] + lq[r] + lk0, -10.f), 10.f);
      float a1 = fminf(fmaxf(s1[r] + lq[r] + lk1, -10.f), 10.f);
      float rm = red16_max(fmaxf(a0, a1));
      float nm = fmaxf(mrow[r], rm);
      float corr = __expf(mrow[r] - nm);
      mrow[r] = nm;
      float p0 = __expf(a0 - nm);
      float p1 = __expf(a1 - nm);
      lrow[r] = lrow[r] * corr + red16_sum(p0 + p1);
      acc0[r] *= corr;
      acc1[r] *= corr;
      int row = r + (half << 3);
      pbuf[row][l15] = (_Float16)p0;
      pbuf[row][16 + l15] = (_Float16)p1;
    }
    __syncthreads();
    // P as A-fragment (16x32, K = kv index within block): two ds_load_b128
    v16h pa = make16(*(const v8h*)(&pbuf[l15][khalf]), *(const v8h*)(&pbuf[l15][16 + khalf]));
    // V B-fragments from transposed layout (K-pairs contiguous): b128 loads
    const _Float16* vr0 = vrow0 + kv0;
    const _Float16* vr1 = vrow1 + kv0;
    v16h vb0 = make16(*(const v8h*)(vr0 + khalf), *(const v8h*)(vr0 + 16 + khalf));
    v16h vb1 = make16(*(const v8h*)(vr1 + khalf), *(const v8h*)(vr1 + 16 + khalf));
    acc0 = wmma16(pa, vb0, acc0);
    acc1 = wmma16(pa, vb1, acc1);
    __syncthreads();
  }
#pragma unroll
  for (int r = 0; r < 8; ++r) {
    int row = qbase + r + (half << 3);
    float inv = 1.f / lrow[r];
    out[(size_t)row * DMODEL + h * HD + l15] = acc0[r] * inv;
    out[(size_t)row * DMODEL + h * HD + 16 + l15] = acc1[r] * inv;
  }
}

// ----------------------------------------------------------------- host
// Input order: recursive dict insertion order from setup_inputs().
enum {
  IN_EG = 0, IN_ED, IN_GLCC, IN_DLCC,
  SG_QKV_W, SG_QKV_B, SG_OUT_W, SG_OUT_B, SG_FF1_W, SG_FF1_B, SG_FF2_W, SG_FF2_B,
  SG_LN1_G, SG_LN1_B, SG_LN2_G, SG_LN2_B,
  SD_QKV_W, SD_QKV_B, SD_OUT_W, SD_OUT_B, SD_FF1_W, SD_FF1_B, SD_FF2_W, SD_FF2_B,
  SD_LN1_G, SD_LN1_B, SD_LN2_G, SD_LN2_B,
  CR_GQ_W, CR_GQ_B, CR_GK_W, CR_GK_B, CR_GV_W, CR_GV_B,
  CR_DQ_W, CR_DQ_B, CR_DK_W, CR_DK_B, CR_DV_W, CR_DV_B,
  CR_OUT_W, CR_OUT_B, CR_GG_W, CR_GG_B, CR_GD_W, CR_GD_B,
  CR_GGLN_G, CR_GGLN_B, CR_GDLN_G, CR_GDLN_B,
  CR_LNG_G, CR_LNG_B, CR_LND_G, CR_LND_B
};

namespace {
struct Arena {
  char* base; size_t off;
  float* f(size_t n) { float* p = (float*)(base + off); off += ((n * 4 + 255) / 256) * 256; return p; }
  _Float16* h(size_t n) { _Float16* p = (_Float16*)(base + off); off += ((n * 2 + 255) / 256) * 256; return p; }
};
inline dim3 g1(size_t n, int b) { return dim3((unsigned)((n + b - 1) / b)); }
}  // namespace

extern "C" void kernel_launch(void* const* d_in, const int* in_sizes, int n_in,
                              void* d_out, int out_size, void* d_ws, size_t ws_size,
                              hipStream_t stream) {
  (void)in_sizes; (void)n_in; (void)out_size; (void)ws_size;
  const float* Eg   = (const float*)d_in[IN_EG];
  const float* Ed   = (const float*)d_in[IN_ED];
  const float* glcc = (const float*)d_in[IN_GLCC];
  const float* dlcc = (const float*)d_in[IN_DLCC];
  auto F = [&](int i) { return (const float*)d_in[i]; };

  Arena ws{(char*)d_ws, 0};

  // --- f16 transposed weights
  _Float16* sg_qkvT = ws.h(128 * 384);
  _Float16* sg_outT = ws.h(128 * 128);
  _Float16* sg_ff1T = ws.h(128 * 512);
  _Float16* sg_ff2T = ws.h(512 * 128);
  _Float16* sd_qkvT = ws.h(128 * 384);
  _Float16* sd_outT = ws.h(128 * 128);
  _Float16* sd_ff1T = ws.h(128 * 512);
  _Float16* sd_ff2T = ws.h(512 * 128);
  _Float16* cr_gqT = ws.h(128 * 128);
  _Float16* cr_gkT = ws.h(128 * 128);
  _Float16* cr_gvT = ws.h(128 * 128);
  _Float16* cr_dqT = ws.h(128 * 128);
  _Float16* cr_dkT = ws.h(128 * 128);
  _Float16* cr_dvT = ws.h(128 * 128);
  _Float16* cr_outT = ws.h(128 * 128);
  _Float16* cr_ggT = ws.h(256 * 128);
  _Float16* cr_gdT = ws.h(256 * 128);

  auto conv = [&](int widx, _Float16* dst, int K, int N) {
    k_w2h_t<<<g1((size_t)K * N, 256), 256, 0, stream>>>(F(widx), dst, K, N);
  };
  conv(SG_QKV_W, sg_qkvT, 128, 384); conv(SG_OUT_W, sg_outT, 128, 128);
  conv(SG_FF1_W, sg_ff1T, 128, 512); conv(SG_FF2_W, sg_ff2T, 512, 128);
  conv(SD_QKV_W, sd_qkvT, 128, 384); conv(SD_OUT_W, sd_outT, 128, 128);
  conv(SD_FF1_W, sd_ff1T, 128, 512); conv(SD_FF2_W, sd_ff2T, 512, 128);
  conv(CR_GQ_W, cr_gqT, 128, 128); conv(CR_GK_W, cr_gkT, 128, 128);
  conv(CR_GV_W, cr_gvT, 128, 128); conv(CR_DQ_W, cr_dqT, 128, 128);
  conv(CR_DK_W, cr_dkT, 128, 128); conv(CR_DV_W, cr_dvT, 128, 128);
  conv(CR_OUT_W, cr_outT, 128, 128);
  conv(CR_GG_W, cr_ggT, 256, 128); conv(CR_GD_W, cr_gdT, 256, 128);

  // --- activation buffers
  float* xg   = ws.f((size_t)NG * 128);
  float* xd   = ws.f((size_t)ND * 128);
  float* nxg  = ws.f((size_t)NG * 128);
  float* nxd  = ws.f((size_t)ND * 128);
  float* qkvg = ws.f((size_t)NG * 384);
  float* qkvd = ws.f((size_t)ND * 384);
  float* hffg = ws.f((size_t)NG * 512);
  float* hffd = ws.f((size_t)ND * 512);
  float* ctxg = ws.f((size_t)NG * 128);
  float* ctxd = ws.f((size_t)ND * 128);
  _Float16* qhg = ws.h((size_t)NG * 128);
  _Float16* khg = ws.h((size_t)NG * 128);
  _Float16* vTg = ws.h((size_t)NG * 128);
  _Float16* qhd = ws.h((size_t)ND * 128);
  _Float16* khd = ws.h((size_t)ND * 128);
  _Float16* vTd = ws.h((size_t)ND * 128);

  auto gemm = [&](const float* A, const _Float16* WT, const float* bias, const float* res,
                  float* C, int M, int N, int K, int act) {
    k_gemm_wmma<<<dim3(M / 16, N / 64), 32, 0, stream>>>(A, WT, bias, res, C, M, N, K, act);
  };

  // --- positional encoding
  k_pos_add<<<g1((size_t)NG * 128, 256), 256, 0, stream>>>(Eg, xg, NG);
  k_pos_add<<<g1((size_t)ND * 128, 256), 256, 0, stream>>>(Ed, xd, ND);

  // --- self-attention blocks
  auto self_block = [&](float* x, int S, const float* lcc, const _Float16* qkvT,
                        const _Float16* outT, const _Float16* ff1T, const _Float16* ff2T,
                        int qkv_b, int out_b, int ff1_b, int ff2_b,
                        int ln1g, int ln1b, int ln2g, int ln2b,
                        float* nx, float* qkvbuf, float* hff, float* ctx,
                        _Float16* qh, _Float16* kh, _Float16* vT) {
    k_layernorm<<<S, 128, 0, stream>>>(x, F(ln1g), F(ln1b), nx, 0);
    gemm(nx, qkvT, F(qkv_b), nullptr, qkvbuf, S, 384, 128, 0);
    k_head_relayout<<<S, 128, 0, stream>>>(qkvbuf + 0,   384, qh, S, 0);
    k_head_relayout<<<S, 128, 0, stream>>>(qkvbuf + 128, 384, kh, S, 1);
    k_head_relayout<<<S, 128, 0, stream>>>(qkvbuf + 256, 384, vT, S, 2);
    k_flash_attn<<<dim3(S / 16, NH), 32, 0, stream>>>(qh, kh, vT, lcc, lcc, ctx, S, S);
    gemm(ctx, outT, F(out_b), x, x, S, 128, 128, 0);
    k_layernorm<<<S, 128, 0, stream>>>(x, F(ln2g), F(ln2b), nx, 0);
    gemm(nx, ff1T, F(ff1_b), nullptr, hff, S, 512, 128, 1);
    gemm(hff, ff2T, F(ff2_b), x, x, S, 128, 512, 0);
  };
  self_block(xg, NG, glcc, sg_qkvT, sg_outT, sg_ff1T, sg_ff2T,
             SG_QKV_B, SG_OUT_B, SG_FF1_B, SG_FF2_B,
             SG_LN1_G, SG_LN1_B, SG_LN2_G, SG_LN2_B,
             nxg, qkvg, hffg, ctxg, qhg, khg, vTg);
  self_block(xd, ND, dlcc, sd_qkvT, sd_outT, sd_ff1T, sd_ff2T,
             SD_QKV_B, SD_OUT_B, SD_FF1_B, SD_FF2_B,
             SD_LN1_G, SD_LN1_B, SD_LN2_G, SD_LN2_B,
             nxd, qkvd, hffd, ctxd, qhd, khd, vTd);

  // --- cross-attention block
  float* gq = qkvg;                 // reuse qkv scratch: 3 * S*128 fits
  float* gk = qkvg + (size_t)NG * 128;
  float* gv = qkvg + (size_t)2 * NG * 128;
  float* dq = qkvd;
  float* dk = qkvd + (size_t)ND * 128;
  float* dv = qkvd + (size_t)2 * ND * 128;

  k_layernorm<<<NG, 128, 0, stream>>>(xg, F(CR_LNG_G), F(CR_LNG_B), nxg, 0);
  k_layernorm<<<ND, 128, 0, stream>>>(xd, F(CR_LND_G), F(CR_LND_B), nxd, 0);
  gemm(nxg, cr_gqT, F(CR_GQ_B), nullptr, gq, NG, 128, 128, 0);
  gemm(nxg, cr_gkT, F(CR_GK_B), nullptr, gk, NG, 128, 128, 0);
  gemm(xg,  cr_gvT, F(CR_GV_B), nullptr, gv, NG, 128, 128, 0);  // v from raw gene
  gemm(nxd, cr_dqT, F(CR_DQ_B), nullptr, dq, ND, 128, 128, 0);
  gemm(nxd, cr_dkT, F(CR_DK_B), nullptr, dk, ND, 128, 128, 0);
  gemm(xd,  cr_dvT, F(CR_DV_B), nullptr, dv, ND, 128, 128, 0);  // v from raw drug

  // relayouts (reuse self-attn f16 buffers)
  k_head_relayout<<<NG, 128, 0, stream>>>(gq, 128, qhg, NG, 0);  // gene q
  k_head_relayout<<<NG, 128, 0, stream>>>(gk, 128, khg, NG, 1);  // gene k
  k_head_relayout<<<NG, 128, 0, stream>>>(gv, 128, vTg, NG, 2);  // gene v^T
  k_head_relayout<<<ND, 128, 0, stream>>>(dq, 128, qhd, ND, 0);  // drug q
  k_head_relayout<<<ND, 128, 0, stream>>>(dk, 128, khd, ND, 1);  // drug k
  k_head_relayout<<<ND, 128, 0, stream>>>(dv, 128, vTd, ND, 2);  // drug v^T

  float* gctx = ctxg;
  float* dctx = ctxd;
  // gene attends to drug; drug attends to gene
  k_flash_attn<<<dim3(NG / 16, NH), 32, 0, stream>>>(qhg, khd, vTd, glcc, dlcc, gctx, NG, ND);
  k_flash_attn<<<dim3(ND / 16, NH), 32, 0, stream>>>(qhd, khg, vTg, dlcc, glcc, dctx, ND, NG);

  // gates: sigmoid(LN(concat[ctx, x] @ Wgate + b))
  float* gcat = hffg;   // NG*512 >= NG*256
  float* dcat = hffd;
  float* t1g = nxg;     // ln inputs no longer needed
  float* t1d = nxd;
  float* gateg = gk;    // f32 projection scratch free after relayout
  float* gated = dk;
  float* gproj = gq;
  float* dproj = dq;

  k_concat<<<g1((size_t)NG * 256, 256), 256, 0, stream>>>(gctx, xg, gcat, NG);
  k_concat<<<g1((size_t)ND * 256, 256), 256, 0, stream>>>(dctx, xd, dcat, ND);
  gemm(gcat, cr_ggT, F(CR_GG_B), nullptr, t1g, NG, 128, 256, 0);
  gemm(dcat, cr_gdT, F(CR_GD_B), nullptr, t1d, ND, 128, 256, 0);
  k_layernorm<<<NG, 128, 0, stream>>>(t1g, F(CR_GGLN_G), F(CR_GGLN_B), gateg, 1);
  k_layernorm<<<ND, 128, 0, stream>>>(t1d, F(CR_GDLN_G), F(CR_GDLN_B), gated, 1);

  gemm(gctx, cr_outT, F(CR_OUT_B), nullptr, gproj, NG, 128, 128, 0);
  gemm(dctx, cr_outT, F(CR_OUT_B), nullptr, dproj, ND, 128, 128, 0);

  float* out_g = (float*)d_out;
  float* out_d = out_g + (size_t)NG * 128;
  k_gated_out<<<g1((size_t)NG * 128, 256), 256, 0, stream>>>(xg, gateg, gproj, out_g, NG * 128);
  k_gated_out<<<g1((size_t)ND * 128, 256), 256, 0, stream>>>(xd, gated, dproj, out_d, ND * 128);
}
